// SelfAttention_7112465842706
// MI455X (gfx1250) — matrix operand order repaired and checked by reading, not compiled, not run on hardware
//
#include <hip/hip_runtime.h>
#include <hip/hip_bf16.h>

#define B_   4
#define S_   2048
#define D_   1024
#define H_   16
#define HD_  64
#define MTOT (B_ * S_)   // 8192

typedef __attribute__((ext_vector_type(16))) __bf16 bf16x16;
typedef __attribute__((ext_vector_type(8)))  __bf16 bf16x8;
typedef __attribute__((ext_vector_type(8)))  float  f32x8;
typedef unsigned int v4u __attribute__((ext_vector_type(4)));
typedef int          v8i __attribute__((ext_vector_type(8)));
typedef int          v4i __attribute__((ext_vector_type(4)));

union Frag {
    bf16x16 v;
    bf16x8  h[2];
};

// ---------------------------------------------------------------------------
// Fragment loaders per CDNA5 ISA 7.12.2 (wave32, 16-bit operands)
// A 16x32 (MxK): lanes 0-15 rows M=0..15 hold K = [0..7]∪[16..23];
//                lanes 16-31 same rows hold K = [8..15]∪[24..31].
// B 32x16 (KxN): lane n%16 = column N; lanes 0-15 hold K=[0..7]∪[16..23], lanes 16-31 K=[8..15]∪[24..31].
// ---------------------------------------------------------------------------
__device__ __forceinline__
bf16x16 load_a_frag_g(const __bf16* __restrict__ base, int ld, int row, int k, int lane) {
    Frag f;
    int r   = row + (lane & 15);
    int off = (lane & 16) ? 8 : 0;
    const __bf16* p = base + (size_t)r * ld + k + off;
    f.h[0] = *(const bf16x8*)(p);
    f.h[1] = *(const bf16x8*)(p + 16);
    return f.v;
}

// A fragment from LDS tile with padded row pitch of 72 bf16 (144 B)
__device__ __forceinline__
bf16x16 load_a_frag_lds(const __bf16* base, int row, int k, int lane) {
    Frag f;
    int r   = row + (lane & 15);
    int off = (lane & 16) ? 8 : 0;
    const __bf16* p = base + r * 72 + k + off;
    f.h[0] = *(const bf16x8*)(p);
    f.h[1] = *(const bf16x8*)(p + 16);
    return f.v;
}

// B operand for C = A * W^T where W is row-major [N,K]: B[k][n] = W[n][k]
__device__ __forceinline__
bf16x16 load_b_frag_nt(const __bf16* __restrict__ base, int ld, int col, int k, int lane) {
    Frag f;
    int c   = col + (lane & 15);
    int off = (lane & 16) ? 8 : 0;
    const __bf16* p = base + (size_t)c * ld + k + off;
    f.h[0] = *(const bf16x8*)(p);
    f.h[1] = *(const bf16x8*)(p + 16);
    return f.v;
}

#define WMMA_BF16(a, b, c) \
    __builtin_amdgcn_wmma_f32_16x16x32_bf16(false, (a), false, (b), (short)0, (c), false, false)

// ---------------------------------------------------------------------------
// TDM: 2-D tensor_load_to_lds descriptor (CDNA5 ISA).
// Loads a tile_d1 x tile_d0 (rows x contiguous-elems) bf16 tile into LDS,
// inserting pad_amount DWORDs after every pad_interval DWORDs.
// ---------------------------------------------------------------------------
__device__ __forceinline__
void tdm_load_2d_bf16(const __bf16* gaddr, unsigned lds_byte_off,
                      unsigned tensor_d0, unsigned tensor_d1,
                      unsigned long long d0_stride,
                      unsigned tile_d0, unsigned tile_d1,
                      unsigned pad_interval_code, unsigned pad_amount_code,
                      bool pad_en)
{
    unsigned long long ga = (unsigned long long)(uintptr_t)gaddr;
    v4u g0;
    g0.x = 1u;                                               // count=1, no gather
    g0.y = lds_byte_off;                                     // lds_addr[31:0]
    g0.z = (unsigned)ga;                                     // global_addr[31:0]
    g0.w = (unsigned)((ga >> 32) & 0x1FFFFFFull) | (2u << 30); // addr[56:32], type=2

    unsigned w0 = (1u << 16)                                 // data_size = 2 bytes
                | (pad_en ? (1u << 20) : 0u)                 // pad_enable
                | (pad_interval_code << 22)
                | (pad_amount_code << 25);
    unsigned long long q0 = (unsigned long long)w0
                          | ((unsigned long long)(tensor_d0 & 0xFFFFu) << 48);
    unsigned long long q1 = ((unsigned long long)tensor_d0 >> 16)
                          | ((unsigned long long)tensor_d1 << 16)
                          | ((unsigned long long)tile_d0 << 48);
    unsigned long long q2 = (unsigned long long)tile_d1
                          | ((d0_stride & 0xFFFFFFFFull) << 32);
    unsigned long long q3 = (d0_stride >> 32) & 0xFFFFull;

    v8i g1;
    g1[0] = (int)(unsigned)q0;  g1[1] = (int)(unsigned)(q0 >> 32);
    g1[2] = (int)(unsigned)q1;  g1[3] = (int)(unsigned)(q1 >> 32);
    g1[4] = (int)(unsigned)q2;  g1[5] = (int)(unsigned)(q2 >> 32);
    g1[6] = (int)(unsigned)q3;  g1[7] = (int)(unsigned)(q3 >> 32);

    v4i z4 = {0, 0, 0, 0};
#if defined(__clang_major__) && (__clang_major__ >= 23)
    v8i z8 = {0, 0, 0, 0, 0, 0, 0, 0};
    __builtin_amdgcn_tensor_load_to_lds(g0, g1, z4, z4, z8, 0);
#else
    __builtin_amdgcn_tensor_load_to_lds(g0, g1, z4, z4, 0);
#endif
}

// ---------------------------------------------------------------------------
// fp32 -> bf16 conversion
// ---------------------------------------------------------------------------
__global__ void f32_to_bf16_kernel(const float* __restrict__ in,
                                   __bf16* __restrict__ out, int n) {
    int i = blockIdx.x * blockDim.x + threadIdx.x;
    if (i < n) out[i] = (__bf16)in[i];
}

// ---------------------------------------------------------------------------
// TDM-staged NT GEMM: C[M,N] = A[M,K] @ W[N,K]^T.
// Workgroup = 256 thr (8 waves, 4M x 2N), tile 128x128; wave tile 32x64.
// A tile (128 x 64 bf16) double-buffered in LDS via tensor_load_to_lds with
// descriptor padding (16 B per 128 B row -> 144 B pitch, conflict-reduced).
// OUT_MODE 0: bf16 row-major [M,N]
// OUT_MODE 1: bf16 V-transpose -> Vt[(b*H+h)*HD + hd][S]
// OUT_MODE 2: f32 row-major + bias
// ---------------------------------------------------------------------------
#define BM 128
#define BN 128
#define KSTAGE 64

template <int OUT_MODE>
__global__ void __launch_bounds__(256)
gemm_tdm_nt_kernel(const __bf16* __restrict__ A, const __bf16* __restrict__ W,
                   void* __restrict__ Cout, const float* __restrict__ bias,
                   int M, int N, int K) {
    __shared__ __align__(16) __bf16 Atile[2][BM][72];   // 72 = 64 + 8 pad elems

    int lane = threadIdx.x & 31;
    int wave = threadIdx.x >> 5;
    int wm = wave >> 1;                    // 0..3
    int wn = wave & 1;                     // 0..1
    int tiles_n = N / BN;
    int bm = (blockIdx.x / tiles_n) * BM;
    int bn = (blockIdx.x % tiles_n) * BN;
    int rowA = wm * 32;                    // wave's rows within LDS tile
    int tn   = bn + wn * 64;               // wave's first output column

    f32x8 acc[2][4];
#pragma unroll
    for (int mi = 0; mi < 2; ++mi)
#pragma unroll
        for (int ni = 0; ni < 4; ++ni) acc[mi][ni] = (f32x8){};

    // prologue: stage K-tile 0 into buffer 0
    if (wave == 0) {
        tdm_load_2d_bf16(A + (size_t)bm * K,
                         (unsigned)(uintptr_t)&Atile[0][0][0],
                         (unsigned)K, (unsigned)M, (unsigned long long)K,
                         KSTAGE, BM,
                         /*pad_interval 32 DW*/ 4u, /*pad 4 DW*/ 3u, true);
    }

    for (int ks = 0; ks < K; ks += KSTAGE) {
        int buf = (ks / KSTAGE) & 1;
        if (wave == 0) {
            if (ks + KSTAGE < K) {
                tdm_load_2d_bf16(A + (size_t)bm * K + (ks + KSTAGE),
                                 (unsigned)(uintptr_t)&Atile[buf ^ 1][0][0],
                                 (unsigned)K, (unsigned)M, (unsigned long long)K,
                                 KSTAGE, BM, 4u, 3u, true);
                __builtin_amdgcn_s_wait_tensorcnt((short)1);  // current tile done
            } else {
                __builtin_amdgcn_s_wait_tensorcnt((short)0);
            }
        }
        __syncthreads();   // tile `buf` visible to all waves

        const __bf16* at = &Atile[buf][0][0];
#pragma unroll
        for (int ki = 0; ki < KSTAGE; ki += 32) {
            bf16x16 a0 = load_a_frag_lds(at, rowA,      ki, lane);
            bf16x16 a1 = load_a_frag_lds(at, rowA + 16, ki, lane);
            int kg = ks + ki;
            bf16x16 b0 = load_b_frag_nt(W, K, tn,      kg, lane);
            bf16x16 b1 = load_b_frag_nt(W, K, tn + 16, kg, lane);
            bf16x16 b2 = load_b_frag_nt(W, K, tn + 32, kg, lane);
            bf16x16 b3 = load_b_frag_nt(W, K, tn + 48, kg, lane);
            acc[0][0] = WMMA_BF16(a0, b0, acc[0][0]);
            acc[0][1] = WMMA_BF16(a0, b1, acc[0][1]);
            acc[0][2] = WMMA_BF16(a0, b2, acc[0][2]);
            acc[0][3] = WMMA_BF16(a0, b3, acc[0][3]);
            acc[1][0] = WMMA_BF16(a1, b0, acc[1][0]);
            acc[1][1] = WMMA_BF16(a1, b1, acc[1][1]);
            acc[1][2] = WMMA_BF16(a1, b2, acc[1][2]);
            acc[1][3] = WMMA_BF16(a1, b3, acc[1][3]);
        }
        __syncthreads();   // done reading `buf` before it is re-staged
    }

    // epilogue: C layout VGPR j -> (M = j + 8*(lane>=16), N = lane%16)
    int ln = lane & 15;
    int hi = (lane >> 4) & 1;
#pragma unroll
    for (int mi = 0; mi < 2; ++mi)
#pragma unroll
        for (int ni = 0; ni < 4; ++ni)
#pragma unroll
            for (int j = 0; j < 8; ++j) {
                int mg = bm + rowA + mi * 16 + j + hi * 8;
                int ng = tn + ni * 16 + ln;
                float v = acc[mi][ni][j];
                if (OUT_MODE == 0) {
                    ((__bf16*)Cout)[(size_t)mg * N + ng] = (__bf16)v;
                } else if (OUT_MODE == 1) {
                    int b  = mg / S_,  s  = mg % S_;
                    int h  = ng / HD_, hd = ng % HD_;
                    ((__bf16*)Cout)[((size_t)(b * H_ + h) * HD_ + hd) * S_ + s] = (__bf16)v;
                } else {
                    ((float*)Cout)[(size_t)mg * N + ng] = v + bias[ng];
                }
            }
}

// ---------------------------------------------------------------------------
// Flash attention: one wave = 16 query rows of one (b,h). Key tiles of 32.
// ---------------------------------------------------------------------------
__global__ void __launch_bounds__(128)
attn_kernel(const __bf16* __restrict__ Q, const __bf16* __restrict__ Kc,
            const __bf16* __restrict__ Vt, __bf16* __restrict__ O) {
    __shared__ __align__(16) __bf16 lds_p[4][16][40];

    int lane = threadIdx.x & 31;
    int wave = threadIdx.x >> 5;
    int gw = blockIdx.x * 4 + wave;
    int qt = gw % (S_ / 16);
    int h  = (gw / (S_ / 16)) % H_;
    int b  = gw / ((S_ / 16) * H_);
    int q0 = qt * 16;

    const __bf16* Qbase = Q  + (size_t)b * S_ * D_ + (size_t)h * HD_;
    const __bf16* Kbase = Kc + (size_t)b * S_ * D_ + (size_t)h * HD_;
    const __bf16* Vbase = Vt + (size_t)(b * H_ + h) * HD_ * S_;

    bf16x16 aQ0 = load_a_frag_g(Qbase, D_, q0,  0, lane);
    bf16x16 aQ1 = load_a_frag_g(Qbase, D_, q0, 32, lane);

    f32x8 o0 = {}, o1 = {}, o2 = {}, o3 = {};
    float mrow[8], lrow[8];
#pragma unroll
    for (int j = 0; j < 8; ++j) { mrow[j] = -1e30f; lrow[j] = 0.0f; }

    int ln = lane & 15;
    int hi = (lane >> 4) & 1;
    const float scale = 0.125f;   // 1/sqrt(64)

    for (int kt = 0; kt < S_; kt += 32) {
        f32x8 sc0 = {}, sc1 = {};
        bf16x16 bk00 = load_b_frag_nt(Kbase, D_, kt,       0, lane);
        bf16x16 bk01 = load_b_frag_nt(Kbase, D_, kt,      32, lane);
        bf16x16 bk10 = load_b_frag_nt(Kbase, D_, kt + 16,  0, lane);
        bf16x16 bk11 = load_b_frag_nt(Kbase, D_, kt + 16, 32, lane);
        sc0 = WMMA_BF16(aQ0, bk00, sc0);
        sc0 = WMMA_BF16(aQ1, bk01, sc0);
        sc1 = WMMA_BF16(aQ0, bk10, sc1);
        sc1 = WMMA_BF16(aQ1, bk11, sc1);

#pragma unroll
        for (int j = 0; j < 8; ++j) {
            float s0 = sc0[j] * scale;
            float s1 = sc1[j] * scale;
            float tmax = fmaxf(s0, s1);
#pragma unroll
            for (int m = 1; m < 16; m <<= 1) tmax = fmaxf(tmax, __shfl_xor(tmax, m, 32));
            float mnew = fmaxf(mrow[j], tmax);
            float corr = __expf(mrow[j] - mnew);
            float p0 = __expf(s0 - mnew);
            float p1 = __expf(s1 - mnew);
            float psum = p0 + p1;
#pragma unroll
            for (int m = 1; m < 16; m <<= 1) psum += __shfl_xor(psum, m, 32);
            lrow[j] = lrow[j] * corr + psum;
            mrow[j] = mnew;
            o0[j] *= corr; o1[j] *= corr; o2[j] *= corr; o3[j] *= corr;
            lds_p[wave][j + hi * 8][ln]      = (__bf16)p0;
            lds_p[wave][j + hi * 8][16 + ln] = (__bf16)p1;
        }
        __builtin_amdgcn_wave_barrier();   // DS in-order per wave; fence scheduler

        Frag pf;
        {
            int off = (lane & 16) ? 8 : 0;
            pf.h[0] = *(const bf16x8*)&lds_p[wave][ln][off];
            pf.h[1] = *(const bf16x8*)&lds_p[wave][ln][16 + off];
        }
        __builtin_amdgcn_wave_barrier();

        bf16x16 bv0 = load_b_frag_nt(Vbase, S_,  0, kt, lane);
        bf16x16 bv1 = load_b_frag_nt(Vbase, S_, 16, kt, lane);
        bf16x16 bv2 = load_b_frag_nt(Vbase, S_, 32, kt, lane);
        bf16x16 bv3 = load_b_frag_nt(Vbase, S_, 48, kt, lane);
        o0 = WMMA_BF16(pf.v, bv0, o0);
        o1 = WMMA_BF16(pf.v, bv1, o1);
        o2 = WMMA_BF16(pf.v, bv2, o2);
        o3 = WMMA_BF16(pf.v, bv3, o3);
    }

    __bf16* Obase = O + (size_t)b * S_ * D_ + (size_t)h * HD_;
#pragma unroll
    for (int j = 0; j < 8; ++j) {
        float inv = 1.0f / lrow[j];
        size_t r = (size_t)(q0 + j + hi * 8) * D_;
        Obase[r +  0 + ln] = (__bf16)(o0[j] * inv);
        Obase[r + 16 + ln] = (__bf16)(o1[j] * inv);
        Obase[r + 32 + ln] = (__bf16)(o2[j] * inv);
        Obase[r + 48 + ln] = (__bf16)(o3[j] * inv);
    }
}

// ---------------------------------------------------------------------------
// Launch
// ---------------------------------------------------------------------------
extern "C" void kernel_launch(void* const* d_in, const int* in_sizes, int n_in,
                              void* d_out, int out_size, void* d_ws, size_t ws_size,
                              hipStream_t stream) {
    const float* x  = (const float*)d_in[0];
    const float* Wq = (const float*)d_in[1];
    const float* Wk = (const float*)d_in[2];
    const float* Wv = (const float*)d_in[3];
    const float* Wo = (const float*)d_in[4];
    const float* bo = (const float*)d_in[5];

    const int n_x = MTOT * D_;     // 8,388,608
    const int n_w = D_ * D_;       // 1,048,576

    __bf16* p = (__bf16*)d_ws;
    __bf16* xb  = p; p += n_x;
    __bf16* wqb = p; p += n_w;
    __bf16* wkb = p; p += n_w;
    __bf16* wvb = p; p += n_w;
    __bf16* wob = p; p += n_w;
    __bf16* Qb  = p; p += n_x;
    __bf16* Kb  = p; p += n_x;
    __bf16* Vtb = p; p += n_x;     // [B*H, HD, S]
    __bf16* AOb = p; p += n_x;

    const int cthr = 256;
    f32_to_bf16_kernel<<<(n_x + cthr - 1) / cthr, cthr, 0, stream>>>(x,  xb,  n_x);
    f32_to_bf16_kernel<<<(n_w + cthr - 1) / cthr, cthr, 0, stream>>>(Wq, wqb, n_w);
    f32_to_bf16_kernel<<<(n_w + cthr - 1) / cthr, cthr, 0, stream>>>(Wk, wkb, n_w);
    f32_to_bf16_kernel<<<(n_w + cthr - 1) / cthr, cthr, 0, stream>>>(Wv, wvb, n_w);
    f32_to_bf16_kernel<<<(n_w + cthr - 1) / cthr, cthr, 0, stream>>>(Wo, wob, n_w);

    const int gemm_blocks = (MTOT / BM) * (D_ / BN);       // 64 * 8 = 512
    gemm_tdm_nt_kernel<0><<<gemm_blocks, 256, 0, stream>>>(xb, wqb, Qb,  nullptr, MTOT, D_, D_);
    gemm_tdm_nt_kernel<0><<<gemm_blocks, 256, 0, stream>>>(xb, wkb, Kb,  nullptr, MTOT, D_, D_);
    gemm_tdm_nt_kernel<1><<<gemm_blocks, 256, 0, stream>>>(xb, wvb, Vtb, nullptr, MTOT, D_, D_);

    const int attn_blocks = (B_ * H_ * (S_ / 16)) / 4;     // 2048
    attn_kernel<<<attn_blocks, 128, 0, stream>>>(Qb, Kb, Vtb, AOb);

    gemm_tdm_nt_kernel<2><<<gemm_blocks, 256, 0, stream>>>(AOb, wob, (void*)d_out, bo,
                                                           MTOT, D_, D_);
}
